// AttentionLayer_88295937671804
// MI455X (gfx1250) — matrix operand order repaired and checked by reading, not compiled, not run on hardware
//
#include <hip/hip_runtime.h>
#include <hip/hip_bf16.h>
#include <math.h>

// Problem sizes (compile-time constants)
#define BB   32
#define SS   2048
#define HH   1024
#define TWOH 2048

typedef __attribute__((ext_vector_type(16))) __bf16 v16bf;
typedef __attribute__((ext_vector_type(8)))  __bf16 v8bf;
typedef __attribute__((ext_vector_type(4)))  __bf16 v4bf;
typedef __attribute__((ext_vector_type(8)))  float  v8f;

// ---------------------------------------------------------------------------
// Kernel A: transpose+convert W_attn rows [H, 3H) (f32, row-major KxN) into
// Wt (bf16, [N][K] = [1024][2048]) so B-matrix fragments are two contiguous 16B runs.
// ---------------------------------------------------------------------------
__global__ void wt_convert_kernel(const float* __restrict__ W, __bf16* __restrict__ Wt) {
    __shared__ float tile[32][33];
    const int n0 = blockIdx.x * 32;   // N tile
    const int k0 = blockIdx.y * 32;   // K tile
    const int tx = threadIdx.x;       // 0..31
    const int ty = threadIdx.y;       // 0..7
    #pragma unroll
    for (int i = 0; i < 32; i += 8) {
        int k = k0 + ty + i;
        tile[ty + i][tx] = W[(size_t)(HH + k) * HH + (n0 + tx)];
    }
    __syncthreads();
    #pragma unroll
    for (int i = 0; i < 32; i += 8) {
        int n = n0 + ty + i;
        Wt[(size_t)n * TWOH + (k0 + tx)] = (__bf16)tile[tx][ty + i];
    }
}

// ---------------------------------------------------------------------------
// Kernel B: h_proj[b][h] = b_attn[h] + sum_k hidden[b][k] * W_attn[k][h]
// (k in [0,H); tiny GEMM, f32 VALU is fine)
// ---------------------------------------------------------------------------
__global__ void hproj_kernel(const float* __restrict__ hidden,
                             const float* __restrict__ W,
                             const float* __restrict__ b_attn,
                             float* __restrict__ hproj) {
    const int h = blockIdx.x * 256 + threadIdx.x;  // < 1024
    const int b = blockIdx.y;
    const float* hid = hidden + (size_t)b * HH;
    float acc = b_attn[h];
    #pragma unroll 8
    for (int k = 0; k < HH; ++k)
        acc = fmaf(hid[k], W[(size_t)k * HH + h], acc);
    hproj[(size_t)b * HH + h] = acc;
}

// ---------------------------------------------------------------------------
// Kernel C: main WMMA kernel.
// Block = 256 threads (8 waves), owns 32 rows (s0..s0+31) of batch b.
// Computes scores[b][s] = sum_n tanh(enc_proj + h_proj)[n] * w_v[n]  (+ b_v)
// enc_proj accumulated with v_wmma_f32_16x16x32_bf16 over K=2048.
// ---------------------------------------------------------------------------
__global__ void __launch_bounds__(256)
scores_kernel(const float* __restrict__ enc,
              const __bf16* __restrict__ Wt,
              const float* __restrict__ hproj,
              const float* __restrict__ w_v,
              const float* __restrict__ b_v,
              float* __restrict__ scores) {
    __shared__ __attribute__((aligned(16))) __bf16 Alds[32 * 512];  // 32 KB
    __shared__ float slds[32];

    const int b    = blockIdx.y;
    const int s0   = blockIdx.x * 32;
    const int tid  = threadIdx.x;
    const int lane = tid & 31;
    const int wave = tid >> 5;          // 0..7, owns n-tiles wave*8 .. wave*8+7
    const int laneHi = lane >> 4;       // 0 | 1
    const int ln     = lane & 15;

    if (tid < 32) slds[tid] = 0.0f;

    const size_t encBase = ((size_t)b * SS + s0) * TWOH;

    v8f acc[2][8];
    #pragma unroll
    for (int m = 0; m < 2; ++m)
        #pragma unroll
        for (int t = 0; t < 8; ++t)
            acc[m][t] = {};

    for (int kh = 0; kh < TWOH; kh += 512) {
        __syncthreads();
        // Stage A tile (32 rows x 512 K-cols) as bf16 into LDS.
        // Vectorized: float4 global loads -> packed 4x bf16 LDS stores.
        for (int i = tid * 4; i < 32 * 512; i += 256 * 4) {
            int row = i >> 9;
            int d   = i & 511;
            const float4 f = *(const float4*)(enc + encBase + (size_t)row * TWOH + kh + d);
            v4bf o;
            o[0] = (__bf16)f.x; o[1] = (__bf16)f.y;
            o[2] = (__bf16)f.z; o[3] = (__bf16)f.w;
            *(v4bf*)(&Alds[i]) = o;
        }
        __syncthreads();

        for (int k0 = 0; k0 < 512; k0 += 32) {
            // A fragments (16-bit A layout: lanes0-15 K={0..7,16..23},
            // lanes16-31 K={8..15,24..31}) for both 16-row sub-tiles.
            const int kb = k0 + laneHi * 8;
            v16bf afr[2];
            #pragma unroll
            for (int m = 0; m < 2; ++m) {
                const __bf16* ap = &Alds[(m * 16 + ln) * 512 + kb];
                v8bf lo = *(const v8bf*)(ap);
                v8bf hi = *(const v8bf*)(ap + 16);
                afr[m] = __builtin_shufflevector(lo, hi,
                          0,1,2,3,4,5,6,7,8,9,10,11,12,13,14,15);
            }
            #pragma unroll
            for (int t = 0; t < 8; ++t) {
                const int n = (wave * 8 + t) * 16 + ln;   // B column
                const __bf16* bp = &Wt[(size_t)n * TWOH + kh + k0 + laneHi * 8];
                __builtin_prefetch(bp + 32, 0, 3);        // next K-chunk, near-scope prefetch
                v16bf bfr = __builtin_shufflevector(*(const v8bf*)bp, *(const v8bf*)(bp + 16),
                          0,1,2,3,4,5,6,7,8,9,10,11,12,13,14,15);  // K={8h..8h+7, 16+8h..16+8h+7}, 16B aligned
                acc[0][t] = __builtin_amdgcn_wmma_f32_16x16x32_bf16(
                    false, afr[0], false, bfr, (short)0, acc[0][t], false, false);
                acc[1][t] = __builtin_amdgcn_wmma_f32_16x16x32_bf16(
                    false, afr[1], false, bfr, (short)0, acc[1][t], false, false);
            }
        }
    }

    // Epilogue: e = tanh(acc + h_proj[b][n]); per-lane partial score = e*w_v[n]
    // C layout: acc[m][t][r] is element (M = m*16 + r + laneHi*8, N = ln + ntile*16).
    float sv[2][8];
    #pragma unroll
    for (int m = 0; m < 2; ++m)
        #pragma unroll
        for (int r = 0; r < 8; ++r) sv[m][r] = 0.0f;

    #pragma unroll
    for (int t = 0; t < 8; ++t) {
        const int n  = (wave * 8 + t) * 16 + ln;
        const float hb = hproj[(size_t)b * HH + n];
        const float wv = w_v[n];
        #pragma unroll
        for (int m = 0; m < 2; ++m)
            #pragma unroll
            for (int r = 0; r < 8; ++r) {
                float e = tanhf(acc[m][t][r] + hb);
                sv[m][r] = fmaf(e, wv, sv[m][r]);
            }
    }

    // Cross-lane/wave reduction into per-row score via LDS float atomics.
    #pragma unroll
    for (int m = 0; m < 2; ++m)
        #pragma unroll
        for (int r = 0; r < 8; ++r) {
            int row = m * 16 + r + laneHi * 8;
            atomicAdd(&slds[row], sv[m][r]);
        }
    __syncthreads();
    if (tid < 32)
        scores[(size_t)b * SS + s0 + tid] = slds[tid] + b_v[0];
}

// ---------------------------------------------------------------------------
// Kernel D: row softmax over S=2048 per batch.
// ---------------------------------------------------------------------------
__global__ void softmax_kernel(const float* __restrict__ scores,
                               float* __restrict__ attn) {
    __shared__ float red[256];
    const int b = blockIdx.x;
    const int tid = threadIdx.x;
    const float* s = scores + (size_t)b * SS;
    float m = -1e30f;
    for (int i = tid; i < SS; i += 256) m = fmaxf(m, s[i]);
    red[tid] = m; __syncthreads();
    for (int off = 128; off > 0; off >>= 1) {
        if (tid < off) red[tid] = fmaxf(red[tid], red[tid + off]);
        __syncthreads();
    }
    m = red[0]; __syncthreads();

    float sum = 0.0f;
    for (int i = tid; i < SS; i += 256) {
        float e = __expf(s[i] - m);
        attn[(size_t)b * SS + i] = e;
        sum += e;
    }
    red[tid] = sum; __syncthreads();
    for (int off = 128; off > 0; off >>= 1) {
        if (tid < off) red[tid] += red[tid + off];
        __syncthreads();
    }
    float inv = 1.0f / red[0];
    for (int i = tid; i < SS; i += 256) attn[(size_t)b * SS + i] *= inv;
}

// ---------------------------------------------------------------------------
// Kernel E: context[b][d] = sum_s attn[b][s] * enc[b][s][d]   (d < 2H)
// Coalesced along d; memory bound on enc re-read.
// ---------------------------------------------------------------------------
__global__ void context_kernel(const float* __restrict__ enc,
                               const float* __restrict__ attn,
                               float* __restrict__ ctx) {
    const int b = blockIdx.y;
    const int d = blockIdx.x * 256 + threadIdx.x;   // < 2048
    const float* e = enc + (size_t)b * SS * TWOH + d;
    const float* a = attn + (size_t)b * SS;
    float c = 0.0f;
    #pragma unroll 4
    for (int s = 0; s < SS; ++s)
        c = fmaf(a[s], e[(size_t)s * TWOH], c);
    ctx[(size_t)b * TWOH + d] = c;
}

// ---------------------------------------------------------------------------
extern "C" void kernel_launch(void* const* d_in, const int* in_sizes, int n_in,
                              void* d_out, int out_size, void* d_ws, size_t ws_size,
                              hipStream_t stream) {
    const float* hidden = (const float*)d_in[0];  // (B,H)
    const float* enc    = (const float*)d_in[1];  // (B,S,2H)
    const float* W      = (const float*)d_in[2];  // (3H,H)
    const float* b_attn = (const float*)d_in[3];  // (H,)
    const float* w_v    = (const float*)d_in[4];  // (H,)
    const float* b_v    = (const float*)d_in[5];  // scalar

    float* ctx  = (float*)d_out;                  // (B,2H) = 65536 floats
    float* attn = (float*)d_out + (size_t)BB * TWOH;  // (B,S) = 65536 floats

    // Workspace layout
    __bf16* Wt    = (__bf16*)d_ws;                                  // 1024*2048 bf16 = 4 MB
    float*  hproj = (float*)((char*)d_ws + (size_t)4 * 1024 * 1024); // 32*1024 f32
    float*  scores = hproj + (size_t)BB * HH;                        // 32*2048 f32

    wt_convert_kernel<<<dim3(HH / 32, TWOH / 32), dim3(32, 8), 0, stream>>>(W, Wt);
    hproj_kernel<<<dim3(HH / 256, BB), 256, 0, stream>>>(hidden, W, b_attn, hproj);
    scores_kernel<<<dim3(SS / 32, BB), 256, 0, stream>>>(enc, Wt, hproj, w_v, b_v, scores);
    softmax_kernel<<<BB, 256, 0, stream>>>(scores, attn);
    context_kernel<<<dim3(TWOH / 256, BB), 256, 0, stream>>>(enc, attn, ctx);
}
